// Transformer_88381837017794
// MI455X (gfx1250) — matrix operand order repaired and checked by reading, not compiled, not run on hardware
//
#include <hip/hip_runtime.h>

// ---------------- model constants ----------------
#define kB   4
#define kS   512
#define kT   512
#define kD   512
#define kH   8
#define kDK  64
#define kFF  2048
#define kV   32000
#define kL   6
#define kBS  (kB * kS)

typedef __attribute__((ext_vector_type(16))) __bf16 v16bf;
typedef __attribute__((ext_vector_type(8)))  __bf16 v8bf;
typedef __attribute__((ext_vector_type(4)))  __bf16 v4bf;
typedef __attribute__((ext_vector_type(8)))  float  v8f;
typedef __attribute__((ext_vector_type(4)))  unsigned int u32x4;
typedef __attribute__((ext_vector_type(8)))  int    i32x8;
typedef __attribute__((ext_vector_type(4)))  int    i32x4;

union Frag16 { v16bf v; v8bf h[2]; };

#define HAS_TDM (__has_builtin(__builtin_amdgcn_tensor_load_to_lds) && \
                 __has_builtin(__builtin_amdgcn_s_wait_tensorcnt))

// =====================================================================
// Tensor Data Mover: load a 64-row x 32-col bf16 tile (row stride
// `strideElems`) into LDS at byte offset ldsOff, padding each 64B row
// with 16B so LDS rows are 40 bf16 (bank-conflict skew). D# per ISA
// CDNA5 ISA. amdgpu-toolchain 6-arg builtin.
// =====================================================================
#if HAS_TDM
__device__ __forceinline__ void tdm_load_tile(unsigned ldsOff, const __bf16* g, long strideElems)
{
    unsigned long long ga = (unsigned long long)g;
    u32x4 g0;
    g0[0] = 1u;                                    // count=1 (valid user descriptor)
    g0[1] = ldsOff;                                // lds_addr (bytes)
    g0[2] = (unsigned)ga;                          // global_addr[31:0]
    g0[3] = (unsigned)((ga >> 32) & 0x01FFFFFFull) | (2u << 30); // addr[56:32] | type=2
    i32x8 g1;
    g1[0] = (1 << 16)     // data_size = 1 -> 2 bytes
          | (1 << 20)     // pad_enable
          | (3 << 22)     // pad_interval: 16 DWORDs (one 64B tile row)
          | (3 << 25);    // pad_amount: 4 DWORDs (16B) -> 80B LDS row pitch
    g1[1] = (int)(32u << 16);                      // tensor_dim0 = 32 (lo16)
    g1[2] = (int)(64u << 16);                      // tensor_dim1 = 64 (lo16 in [31:16])
    g1[3] = (int)(32u << 16);                      // tile_dim0 = 32
    g1[4] = 64;                                    // tile_dim1 = 64, tile_dim2 = 0
    g1[5] = (int)(unsigned)strideElems;            // tensor_dim0_stride[31:0]
    g1[6] = (int)(unsigned)((unsigned long long)strideElems >> 32);
    g1[7] = 0;
    i32x4 zz4 = {0, 0, 0, 0};
    i32x8 zz8 = {0, 0, 0, 0, 0, 0, 0, 0};
    __builtin_amdgcn_tensor_load_to_lds(g0, g1, zz4, zz4, zz8, 0);
}
#endif

// stage a 64x32 bf16 tile into padded LDS [64][40]
__device__ __forceinline__ void stage_bf16_tile(__bf16 (*dst)[40], const __bf16* g,
                                                long strideElems, int tid)
{
#if HAS_TDM
    if ((tid >> 5) == 0)   // one wave issues the DMA (EXEC ignored by TDM)
        tdm_load_tile((unsigned)(unsigned long long)&dst[0][0], g, strideElems);
#else
    #pragma unroll
    for (int it = 0; it < 2; ++it) {
        int c = tid + it * 128;            // 256 chunks of 16B
        int row = c >> 2, ch = c & 3;
        *(uint4*)&dst[row][ch * 8] = *(const uint4*)&g[(long)row * strideElems + ch * 8];
    }
#endif
}

__device__ __forceinline__ void tile_wait()
{
#if HAS_TDM
    __builtin_amdgcn_s_wait_tensorcnt(0);
#endif
}

// =====================================================================
// Generic batched WMMA GEMM. A: bf16, k-contiguous, row stride sAm.
// B modes:
//   BM_W_F32 : f32 weights, B(k,n)=B[k*ldb+n]      (n contiguous)
//   BM_K_BF16: bf16,        B(k,n)=B[n*ldb+k]      (k contiguous, TDM)
//   BM_N_BF16: bf16,        B(k,n)=B[k*ldb+n]      (n contiguous)
// Outputs: optional f32 C (bias/resid/relu/scale) and/or bf16 Cbf.
// Batch z: zb=z/nH, zh=z%nH; offsets off = zb*OffB + zh*OffH.
// 128 threads = 4 waves; each wave owns a 32x32 quadrant of 64x64 tile.
// =====================================================================
enum { BM_W_F32 = 0, BM_K_BF16 = 1, BM_N_BF16 = 2 };

template <int BMODE>
__global__ __launch_bounds__(128) void wmma_gemm_kernel(
    const __bf16* __restrict__ A, const void* __restrict__ Bv,
    const float* __restrict__ bias, const float* __restrict__ resid,
    float* __restrict__ C, __bf16* __restrict__ Cbf,
    int K, long sAm, long aOffB, long aOffH,
    long ldb, long bOffB, long bOffH,
    long ldc, long cOffB, long cOffH,
    int nH, float scale, int relu)
{
    __shared__ __bf16 lA[64][40];   // [m][k]
    __shared__ __bf16 lB[64][40];   // [n][k] (transposed)

    const int z  = blockIdx.z;
    const int zb = z / nH;
    const int zh = z - zb * nH;
    const int m0 = blockIdx.y * 64;
    const int n0 = blockIdx.x * 64;

    const __bf16* Ab = A + (long)zb * aOffB + (long)zh * aOffH + (long)m0 * sAm;
    const long    bO = (long)zb * bOffB + (long)zh * bOffH;
    const long    cO = (long)zb * cOffB + (long)zh * cOffH;

    const int tid  = threadIdx.x;
    const int lane = tid & 31;
    const int wave = tid >> 5;
    const int wr   = (wave >> 1) * 32;
    const int wc   = (wave & 1) * 32;
    const int half = lane >> 4;
    const int ln   = lane & 15;

    v8f acc[2][2] = {};

    for (int k0 = 0; k0 < K; k0 += 32) {
        __syncthreads();   // previous tile fully consumed

        // ---- A tile via Tensor Data Mover ----
        stage_bf16_tile(lA, Ab + k0, sAm, tid);

        // ---- B tile ----
        if (BMODE == BM_K_BF16) {
            const __bf16* Bb = (const __bf16*)Bv + bO + (long)n0 * ldb;
            stage_bf16_tile(lB, Bb + k0, ldb, tid);
        } else if (BMODE == BM_W_F32) {
            const float* Bb = (const float*)Bv + bO;
            #pragma unroll
            for (int it = 0; it < 4; ++it) {
                int j  = tid + it * 128;       // 512 float4 chunks
                int k  = j >> 4;
                int n4 = (j & 15) * 4;
                float4 f = *(const float4*)&Bb[(long)(k0 + k) * ldb + (n0 + n4)];
                lB[n4 + 0][k] = (__bf16)f.x;
                lB[n4 + 1][k] = (__bf16)f.y;
                lB[n4 + 2][k] = (__bf16)f.z;
                lB[n4 + 3][k] = (__bf16)f.w;
            }
            if (k0 + 32 < K)
                __builtin_prefetch(&Bb[(long)(k0 + 32 + (tid >> 4)) * ldb + n0], 0, 1);
        } else { // BM_N_BF16
            const __bf16* Bb = (const __bf16*)Bv + bO;
            #pragma unroll
            for (int it = 0; it < 4; ++it) {
                int j  = tid + it * 128;
                int k  = j >> 4;
                int n4 = (j & 15) * 4;
                v4bf f = *(const v4bf*)&Bb[(long)(k0 + k) * ldb + (n0 + n4)];
                lB[n4 + 0][k] = f[0];
                lB[n4 + 1][k] = f[1];
                lB[n4 + 2][k] = f[2];
                lB[n4 + 3][k] = f[3];
            }
        }

        tile_wait();        // TENSORcnt == 0 (TDM writes visible)
        __syncthreads();

        // ---- fragments per CDNA5 16-bit VGPR layouts ----
        Frag16 fa[2], fb[2];
        #pragma unroll
        for (int mt = 0; mt < 2; ++mt) {
            int m = wr + mt * 16 + ln;
            fa[mt].h[0] = *(const v8bf*)&lA[m][half * 8];        // K = half*8 + i
            fa[mt].h[1] = *(const v8bf*)&lA[m][16 + half * 8];   // K = 16 + half*8 + i
        }
        #pragma unroll
        for (int nt = 0; nt < 2; ++nt) {
            int n = wc + nt * 16 + ln;
            fb[nt].h[0] = *(const v8bf*)&lB[n][half * 8];        // K = half*8 + i
            fb[nt].h[1] = *(const v8bf*)&lB[n][16 + half * 8];   // K = 16 + half*8 + i
        }
        #pragma unroll
        for (int mt = 0; mt < 2; ++mt)
            #pragma unroll
            for (int nt = 0; nt < 2; ++nt)
                acc[mt][nt] = __builtin_amdgcn_wmma_f32_16x16x32_bf16(
                    false, fa[mt].v, false, fb[nt].v,
                    (short)0, acc[mt][nt], false, false);
    }

    // ---- epilogue: D layout M = vgpr + 8*half, N = lane&15 ----
    float*  Cb  = C   ? C   + cO : nullptr;
    __bf16* CbB = Cbf ? Cbf + cO : nullptr;
    const float* Rb = resid ? resid + cO : nullptr;
    #pragma unroll
    for (int mt = 0; mt < 2; ++mt) {
        #pragma unroll
        for (int nt = 0; nt < 2; ++nt) {
            const int n = n0 + wc + nt * 16 + ln;
            const float bval = bias ? bias[n] : 0.0f;
            #pragma unroll
            for (int e = 0; e < 8; ++e) {
                const int m = m0 + wr + mt * 16 + e + half * 8;
                float val = acc[mt][nt][e] * scale + bval;
                if (Rb)   val += Rb[(long)m * ldc + n];
                if (relu) val = fmaxf(val, 0.0f);
                if (Cb)   Cb[(long)m * ldc + n]  = val;
                if (CbB)  CbB[(long)m * ldc + n] = (__bf16)val;
            }
        }
    }
}

// =====================================================================
// Embedding + sinusoidal PE -> f32 and bf16 copies
// =====================================================================
__global__ __launch_bounds__(128) void embed_kernel(
    const int* __restrict__ tok, const float* __restrict__ emb,
    float* __restrict__ out, __bf16* __restrict__ outBf, int seqLen, int D)
{
    const int row = blockIdx.x;
    const int s   = row % seqLen;
    const int t   = tok[row];
    const float nln = -9.210340371976184f;
    for (int d = threadIdx.x; d < D; d += 128) {
        int p = d >> 1;
        float div = __expf(nln * (2.0f * (float)p) / (float)D);
        float ang = (float)s * div;
        float pe  = (d & 1) ? __cosf(ang) : __sinf(ang);
        float v = emb[(long)t * D + d] + pe;
        out[(long)row * D + d]   = v;
        outBf[(long)row * D + d] = (__bf16)v;
    }
}

// =====================================================================
// Masked softmax; reads f32 scores, writes bf16 probabilities.
// mode 0: keep where src token != 0 at key j; mode 1: causal (j<=q)
// =====================================================================
__global__ __launch_bounds__(128) void softmax_kernel(
    float* __restrict__ sc, __bf16* __restrict__ pOut,
    const int* __restrict__ srcTok, int mode, int Sk, int srcLen)
{
    const int q = blockIdx.x, h = blockIdx.y, b = blockIdx.z;
    const long rowOff = (((long)b * gridDim.y + h) * gridDim.x + q) * (long)Sk;
    float*  r  = sc   + rowOff;
    __bf16* po = pOut + rowOff;
    __shared__ float red[128];
    const int tid = threadIdx.x;

    float lm = -3.4e38f;
    for (int j = tid; j < Sk; j += 128) {
        float v = r[j];
        bool keep = (mode == 0) ? (srcTok[b * srcLen + j] != 0) : (j <= q);
        v = keep ? v : -1e9f;
        r[j] = v;
        lm = fmaxf(lm, v);
    }
    red[tid] = lm; __syncthreads();
    for (int s = 64; s > 0; s >>= 1) { if (tid < s) red[tid] = fmaxf(red[tid], red[tid + s]); __syncthreads(); }
    const float mx = red[0]; __syncthreads();

    float ls = 0.0f;
    for (int j = tid; j < Sk; j += 128) { float e = __expf(r[j] - mx); r[j] = e; ls += e; }
    red[tid] = ls; __syncthreads();
    for (int s = 64; s > 0; s >>= 1) { if (tid < s) red[tid] += red[tid + s]; __syncthreads(); }
    const float inv = 1.0f / red[0];
    for (int j = tid; j < Sk; j += 128) po[j] = (__bf16)(r[j] * inv);
}

// =====================================================================
// LayerNorm -> f32 (residual stream) and bf16 (next GEMM A operand)
// =====================================================================
__global__ __launch_bounds__(128) void layernorm_kernel(
    const float* __restrict__ in, const float* __restrict__ g,
    const float* __restrict__ bb, float* __restrict__ out,
    __bf16* __restrict__ outBf, int D)
{
    const long row = (long)blockIdx.x * D;
    const float* x = in + row;
    __shared__ float red[128];
    const int tid = threadIdx.x;

    float s = 0.0f;
    for (int d = tid; d < D; d += 128) s += x[d];
    red[tid] = s; __syncthreads();
    for (int k = 64; k > 0; k >>= 1) { if (tid < k) red[tid] += red[tid + k]; __syncthreads(); }
    const float mean = red[0] / (float)D; __syncthreads();

    float v = 0.0f;
    for (int d = tid; d < D; d += 128) { float t = x[d] - mean; v += t * t; }
    red[tid] = v; __syncthreads();
    for (int k = 64; k > 0; k >>= 1) { if (tid < k) red[tid] += red[tid + k]; __syncthreads(); }
    const float inv = rsqrtf(red[0] / (float)D + 1e-5f);

    for (int d = tid; d < D; d += 128) {
        float o = (x[d] - mean) * inv * g[d] + bb[d];
        out[row + d]   = o;
        outBf[row + d] = (__bf16)o;
    }
}

// =====================================================================
// Host orchestration
// =====================================================================
extern "C" void kernel_launch(void* const* d_in, const int* in_sizes, int n_in,
                              void* d_out, int out_size, void* d_ws, size_t ws_size,
                              hipStream_t stream)
{
    (void)in_sizes; (void)n_in; (void)out_size; (void)ws_size;

    const int*   src     = (const int*)d_in[0];
    const int*   tgt     = (const int*)d_in[1];
    const float* src_emb = (const float*)d_in[2];
    const float* tgt_emb = (const float*)d_in[3];
    const float* ea[10]; for (int i = 0; i < 10; ++i) ea[i]  = (const float*)d_in[4 + i];
    const float* ef[6];  for (int i = 0; i < 6;  ++i) ef[i]  = (const float*)d_in[14 + i];
    const float* dsA[10]; for (int i = 0; i < 10; ++i) dsA[i] = (const float*)d_in[20 + i];
    const float* dcA[10]; for (int i = 0; i < 10; ++i) dcA[i] = (const float*)d_in[30 + i];
    const float* df[6];   for (int i = 0; i < 6;  ++i) df[i]  = (const float*)d_in[40 + i];
    const float* out_w = (const float*)d_in[46];
    const float* out_b = (const float*)d_in[47];
    float* logits = (float*)d_out;

    // -------- workspace (256B-aligned bump allocator) --------
    char* base = (char*)d_ws;
    size_t off = 0;
    auto allocF = [&](long n) { float*  p = (float*) (base + off); off += ((n * 4 + 255) / 256) * 256; return p; };
    auto allocH = [&](long n) { __bf16* p = (__bf16*)(base + off); off += ((n * 2 + 255) / 256) * 256; return p; };

    float*  x    = allocF((long)kBS * kD);
    float*  y    = allocF((long)kBS * kD);
    float*  bT   = allocF((long)kBS * kD);
    float*  Sc   = allocF((long)kB * kH * kS * kS);
    __bf16* xbf  = allocH((long)kBS * kD);
    __bf16* ybf  = allocH((long)kBS * kD);
    __bf16* Qbf  = allocH((long)kBS * kD);
    __bf16* Kbf  = allocH((long)kBS * kD);
    __bf16* Vbf  = allocH((long)kBS * kD);
    __bf16* Obf  = allocH((long)kBS * kD);
    __bf16* Hbf  = allocH((long)kBS * kFF);
    __bf16* Pbf  = allocH((long)kB * kH * kS * kS);

    // weights GEMM: A bf16 @ B f32[K,N], optional f32/bf16 outs
    auto gemmW = [&](const __bf16* A, long sAm, const float* W, const float* bias,
                     const float* resid, float* C, __bf16* Cbf,
                     int M, int N, int K, float scale, int relu) {
        dim3 grid(N / 64, M / 64, 1);
        wmma_gemm_kernel<BM_W_F32><<<grid, dim3(128), 0, stream>>>(
            A, W, bias, resid, C, Cbf, K, sAm, 0, 0,
            (long)N, 0, 0, (long)N, 0, 0, 1, scale, relu);
    };

    // full MHA block; result LN'ed back into (xq, xqbf)
    auto attn = [&](float* xq, __bf16* xqbf, const __bf16* xkvbf,
                    const float* p[10], int layer, int maskMode) {
        const float* Wq = p[0] + (long)layer * kD * kD;  const float* Bq = p[1] + (long)layer * kD;
        const float* Wk = p[2] + (long)layer * kD * kD;  const float* Bk = p[3] + (long)layer * kD;
        const float* Wv = p[4] + (long)layer * kD * kD;  const float* Bv = p[5] + (long)layer * kD;
        const float* Wo = p[6] + (long)layer * kD * kD;  const float* Bo = p[7] + (long)layer * kD;
        const float* G  = p[8] + (long)layer * kD;       const float* Bb = p[9] + (long)layer * kD;

        gemmW(xqbf,  kD, Wq, Bq, nullptr, nullptr, Qbf, kBS, kD, kD, 1.0f, 0);
        gemmW(xkvbf, kD, Wk, Bk, nullptr, nullptr, Kbf, kBS, kD, kD, 1.0f, 0);
        gemmW(xkvbf, kD, Wv, Bv, nullptr, nullptr, Vbf, kBS, kD, kD, 1.0f, 0);

        // scores = Q.K^T / 8 : both operands bf16 k-contiguous -> dual TDM
        {
            dim3 grid(kS / 64, kS / 64, kB * kH);
            wmma_gemm_kernel<BM_K_BF16><<<grid, dim3(128), 0, stream>>>(
                Qbf, Kbf, nullptr, nullptr, Sc, nullptr, kDK,
                kD, (long)kS * kD, kDK,
                kD, (long)kS * kD, kDK,
                kS, (long)kH * kS * kS, (long)kS * kS,
                kH, 0.125f, 0);
        }
        softmax_kernel<<<dim3(kS, kH, kB), dim3(128), 0, stream>>>(Sc, Pbf, src, maskMode, kS, kS);
        // O = P @ V : A bf16 (TDM), B bf16 n-contiguous
        {
            dim3 grid(kDK / 64, kS / 64, kB * kH);
            wmma_gemm_kernel<BM_N_BF16><<<grid, dim3(128), 0, stream>>>(
                Pbf, Vbf, nullptr, nullptr, nullptr, Obf, kS,
                kS, (long)kH * kS * kS, (long)kS * kS,
                kD, (long)kS * kD, kDK,
                kD, (long)kS * kD, kDK,
                kH, 1.0f, 0);
        }
        gemmW(Obf, kD, Wo, Bo, xq, bT, nullptr, kBS, kD, kD, 1.0f, 0);
        layernorm_kernel<<<dim3(kBS), dim3(128), 0, stream>>>(bT, G, Bb, xq, xqbf, kD);
    };

    auto ffn = [&](float* xio, __bf16* xbf_, const float* p[6], int layer) {
        const float* W1 = p[0] + (long)layer * kD * kFF;  const float* B1 = p[1] + (long)layer * kFF;
        const float* W2 = p[2] + (long)layer * kFF * kD;  const float* B2 = p[3] + (long)layer * kD;
        const float* G  = p[4] + (long)layer * kD;        const float* Bb = p[5] + (long)layer * kD;
        gemmW(xbf_, kD,  W1, B1, nullptr, nullptr, Hbf, kBS, kFF, kD, 1.0f, 1);
        gemmW(Hbf,  kFF, W2, B2, xio,     bT,      nullptr, kBS, kD, kFF, 1.0f, 0);
        layernorm_kernel<<<dim3(kBS), dim3(128), 0, stream>>>(bT, G, Bb, xio, xbf_, kD);
    };

    // -------- encoder --------
    embed_kernel<<<dim3(kBS), dim3(128), 0, stream>>>(src, src_emb, x, xbf, kS, kD);
    for (int i = 0; i < kL; ++i) {
        attn(x, xbf, xbf, ea, i, 0);
        ffn(x, xbf, ef, i);
    }

    // -------- decoder --------
    embed_kernel<<<dim3(kBS), dim3(128), 0, stream>>>(tgt, tgt_emb, y, ybf, kT, kD);
    for (int i = 0; i < kL; ++i) {
        attn(y, ybf, ybf, dsA, i, 1);
        attn(y, ybf, xbf, dcA, i, 0);
        ffn(y, ybf, df, i);
    }

    // -------- output projection --------
    gemmW(ybf, kD, out_w, out_b, nullptr, logits, nullptr, kBS, kV, kD, 1.0f, 0);
}
